// DecoderRNNFeed_30107720745676
// MI455X (gfx1250) — compile-verified, matrix operand order checked by reading, not run on hardware
//
#include <hip/hip_runtime.h>
#include <hip/hip_bf16.h>
#include <math.h>

// Dims (fixed by reference): V=32000, E=1024, H=1024, L=1024, B=64, T=128
#define BB 64
#define TT 128
#define EE 1024
#define HH 1024
#define LL 1024
#define HE 2048
#define H3 3072
// d_out layout: aws [B,T,L] | h_fin [B,H] | aws again [B,T,L]
#define OUT_OFF2 ((size_t)BB * TT * LL)
#define OUT_OFF3 (OUT_OFF2 + (size_t)BB * HH)

typedef __bf16 bf16_t;
typedef __attribute__((ext_vector_type(16))) __bf16 v16bf;
typedef __attribute__((ext_vector_type(8)))  __bf16 v8bf;
typedef __attribute__((ext_vector_type(4)))  __bf16 v4bf;
typedef __attribute__((ext_vector_type(8)))  float  v8f;
typedef __attribute__((ext_vector_type(4)))  float  v4f;

// ---------------------------------------------------------------- utilities
// Vectorized f32 -> bf16 (n must be a multiple of 4; all our sizes are)
__global__ __launch_bounds__(256) void k_f32_to_bf16(const float* __restrict__ src,
                                                     bf16_t* __restrict__ dst, size_t n4) {
  size_t stride = (size_t)gridDim.x * blockDim.x;
  for (size_t i = (size_t)blockIdx.x * blockDim.x + threadIdx.x; i < n4; i += stride) {
    v4f s = *(const v4f*)(src + i * 4);
    v4bf d;
#pragma unroll
    for (int k = 0; k < 4; ++k) d[k] = (bf16_t)s[k];
    *(v4bf*)(dst + i * 4) = d;
  }
}

// cat[b, 0:1024]   = bf16(emb_table[tgt[b,t]])
// cat[b, 1024:2048]= bf16(h[b])
__global__ __launch_bounds__(256) void k_embed_cat(const int* __restrict__ tgt,
                                                   const float* __restrict__ emb,
                                                   const float* __restrict__ h,
                                                   bf16_t* __restrict__ cat, int t) {
  int b = blockIdx.x;
  int j = blockIdx.y * 256 + threadIdx.x;           // 0..2047
  int tok = tgt[b * TT + t];
  float v = (j < EE) ? emb[(size_t)tok * EE + j] : h[(size_t)b * HH + (j - EE)];
  cat[(size_t)b * HE + j] = (bf16_t)v;
}

__global__ __launch_bounds__(256) void k_ctx_to_cat(const float* __restrict__ ctx,
                                                    bf16_t* __restrict__ cat) {
  int b = blockIdx.x;
  int j = blockIdx.y * 256 + threadIdx.x;           // 0..1023
  cat[(size_t)b * HE + EE + j] = (bf16_t)ctx[(size_t)b * HH + j];
}

// ------------------------------------------------------- WMMA bf16 GEMM
// out[m, n] = sum_k A[m,k] * W[n,k] + bias[n]   (M fixed = 64)
// A: bf16 row-major, leading dim lda. W: bf16 row-major [N, K] (ldw == K).
// 256-thread blocks = 8 waves; each wave owns one 16-wide N strip and computes
// the full M=64 (4 accumulators), so each B tile load feeds 4 WMMAs.
__global__ __launch_bounds__(256) void k_wmma_gemm(const bf16_t* __restrict__ A, int lda,
                                                   const bf16_t* __restrict__ W, int K,
                                                   const float* __restrict__ bias,
                                                   float* __restrict__ out,
                                                   bf16_t* __restrict__ out_bf,
                                                   int N, int relu) {
  const int wave = threadIdx.x >> 5;
  const int lane = threadIdx.x & 31;
  const int lr   = lane & 15;   // row (A) / col (B) within tile
  const int kh   = lane >> 4;   // K half-select per documented layout
  const int n0 = (blockIdx.x * 8 + wave) * 16;

  const bf16_t* brow = W + (size_t)(n0 + lr) * K;   // B column n0+lr = W row
  const bf16_t* ar0  = A + (size_t)lr * lda;        // m-tile mt adds 16*lda

  v8f c[4] = {};
  for (int k0 = 0; k0 < K; k0 += 32) {
    const int off = k0 + kh * 8;  // i=0..7 -> K=off+i ; i=8..15 -> K=off+16+(i-8)
    v8bf blo = *(const v8bf*)(brow + off);
    v8bf bhi = *(const v8bf*)(brow + off + 16);
    v16bf b;
#pragma unroll
    for (int i = 0; i < 8; ++i) { b[i] = blo[i]; b[i + 8] = bhi[i]; }
#pragma unroll
    for (int mt = 0; mt < 4; ++mt) {
      const bf16_t* ap = ar0 + (size_t)mt * 16 * lda + off;
      v8bf alo = *(const v8bf*)(ap);
      v8bf ahi = *(const v8bf*)(ap + 16);
      v16bf a;
#pragma unroll
      for (int i = 0; i < 8; ++i) { a[i] = alo[i]; a[i + 8] = ahi[i]; }
      c[mt] = __builtin_amdgcn_wmma_f32_16x16x32_bf16(false, a, false, b, (short)0,
                                                      c[mt], false, false);
    }
  }

  // C/D layout: VGPR r -> M = mt*16 + kh*8 + r ; N = n0 + lr
  const int ocol = n0 + lr;
  const float bv = bias ? bias[ocol] : 0.0f;
#pragma unroll
  for (int mt = 0; mt < 4; ++mt) {
    const int orow0 = mt * 16 + kh * 8;
#pragma unroll
    for (int r = 0; r < 8; ++r) {
      float v = c[mt][r] + bv;
      if (relu) v = fmaxf(v, 0.0f);
      size_t idx = (size_t)(orow0 + r) * N + ocol;
      out[idx] = v;
      if (out_bf) out_bf[idx] = (bf16_t)v;
    }
  }
}

// ------------------------------------------------- softmax + output writes
__global__ __launch_bounds__(256) void k_softmax_write(const float* __restrict__ logits,
                                                       float* __restrict__ aw,
                                                       float* __restrict__ out, int t) {
  __shared__ float red[256];
  const int b = blockIdx.x;
  const float* row = logits + (size_t)b * LL;
  float m = -3.4e38f;
  for (int i = threadIdx.x; i < LL; i += 256) m = fmaxf(m, row[i]);
  red[threadIdx.x] = m; __syncthreads();
  for (int s = 128; s > 0; s >>= 1) {
    if (threadIdx.x < s) red[threadIdx.x] = fmaxf(red[threadIdx.x], red[threadIdx.x + s]);
    __syncthreads();
  }
  m = red[0]; __syncthreads();
  float sum = 0.0f;
  for (int i = threadIdx.x; i < LL; i += 256) {
    float e = expf(row[i] - m);
    aw[(size_t)b * LL + i] = e;
    sum += e;
  }
  red[threadIdx.x] = sum; __syncthreads();
  for (int s = 128; s > 0; s >>= 1) {
    if (threadIdx.x < s) red[threadIdx.x] += red[threadIdx.x + s];
    __syncthreads();
  }
  const float inv = 1.0f / red[0];
  const size_t o1 = ((size_t)b * TT + t) * LL;
  for (int i = threadIdx.x; i < LL; i += 256) {
    float v = aw[(size_t)b * LL + i] * inv;
    aw[(size_t)b * LL + i] = v;
    out[o1 + i] = v;
    out[OUT_OFF3 + o1 + i] = v;
  }
}

// --------------------- context: ctx[b,h] = sum_l aw[b,l]*enc[b,l,h] ---------
// 128 threads per block; thread owns 8 h-values (16B bf16 load / lane), so each
// iteration reads one full 2KB row of enc[b,l,:] coalesced (L2-resident bf16).
__global__ __launch_bounds__(128) void k_ctx_bf16(const float* __restrict__ aw,
                                                  const bf16_t* __restrict__ enc,
                                                  float* __restrict__ ctx) {
  __shared__ float saw[LL];
  const int b = blockIdx.x;
  for (int i = threadIdx.x; i < LL; i += 128) saw[i] = aw[(size_t)b * LL + i];
  __syncthreads();
  const bf16_t* eb = enc + (size_t)b * LL * HH + threadIdx.x * 8;
  float acc[8] = {0.f, 0.f, 0.f, 0.f, 0.f, 0.f, 0.f, 0.f};
  for (int l = 0; l < LL; ++l) {
    if (l + 16 < LL)
      __builtin_prefetch(eb + (size_t)(l + 16) * HH, 0, 0);  // global_prefetch_b8
    v8bf e = *(const v8bf*)(eb + (size_t)l * HH);
    float w = saw[l];
#pragma unroll
    for (int i = 0; i < 8; ++i) acc[i] += w * (float)e[i];
  }
#pragma unroll
  for (int i = 0; i < 8; ++i)
    ctx[(size_t)b * HH + threadIdx.x * 8 + i] = acc[i];
}

__global__ __launch_bounds__(128) void k_ctx_f32(const float* __restrict__ aw,
                                                 const float* __restrict__ enc,
                                                 float* __restrict__ ctx) {
  __shared__ float saw[LL];
  const int b = blockIdx.x;
  for (int i = threadIdx.x; i < LL; i += 128) saw[i] = aw[(size_t)b * LL + i];
  __syncthreads();
  const float* eb = enc + (size_t)b * LL * HH + threadIdx.x * 8;
  float acc[8] = {0.f, 0.f, 0.f, 0.f, 0.f, 0.f, 0.f, 0.f};
  for (int l = 0; l < LL; ++l) {
    v4f e0 = *(const v4f*)(eb + (size_t)l * HH);
    v4f e1 = *(const v4f*)(eb + (size_t)l * HH + 4);
    float w = saw[l];
#pragma unroll
    for (int i = 0; i < 4; ++i) { acc[i] += w * e0[i]; acc[i + 4] += w * e1[i]; }
  }
#pragma unroll
  for (int i = 0; i < 8; ++i)
    ctx[(size_t)b * HH + threadIdx.x * 8 + i] = acc[i];
}

// ----------------------------------------------------------- GRU elementwise
__global__ __launch_bounds__(256) void k_gru(const float* __restrict__ gi,
                                             const float* __restrict__ gh,
                                             const float* __restrict__ h,
                                             float* __restrict__ hn) {
  const int b = blockIdx.x;
  const int j = blockIdx.y * 256 + threadIdx.x;
  const size_t base = (size_t)b * H3;
  float ir = gi[base + j], iz = gi[base + HH + j], in_ = gi[base + 2 * HH + j];
  float hr = gh[base + j], hz = gh[base + HH + j], hn_ = gh[base + 2 * HH + j];
  float r = 1.0f / (1.0f + expf(-(ir + hr)));
  float z = 1.0f / (1.0f + expf(-(iz + hz)));
  float n = tanhf(in_ + r * hn_);
  float hv = h[(size_t)b * HH + j];
  hn[(size_t)b * HH + j] = (1.0f - z) * n + z * hv;
}

// =========================================================== host launcher
extern "C" void kernel_launch(void* const* d_in, const int* in_sizes, int n_in,
                              void* d_out, int out_size, void* d_ws, size_t ws_size,
                              hipStream_t stream) {
  const int*   tgt     = (const int*)  d_in[0];
  const float* hidden  = (const float*)d_in[1];
  const float* enc     = (const float*)d_in[2];
  // d_in[3] = batch_size scalar (unused; B fixed)
  const float* emb     = (const float*)d_in[4];
  const float* attn_W  = (const float*)d_in[5];
  const float* attn_b  = (const float*)d_in[6];
  const float* comb_W  = (const float*)d_in[7];
  const float* comb_b  = (const float*)d_in[8];
  const float* W_ih    = (const float*)d_in[9];
  const float* b_ih    = (const float*)d_in[10];
  const float* W_hh    = (const float*)d_in[11];
  const float* b_hh    = (const float*)d_in[12];
  float* out = (float*)d_out;

  char* ws = (char*)d_ws;
  size_t cur = 0;
  auto alloc = [&](size_t bytes) -> char* {
    char* p = ws + cur;
    cur += (bytes + 255) & ~(size_t)255;
    return p;
  };

  bf16_t* attnWb = (bf16_t*)alloc((size_t)LL * HE * 2);
  bf16_t* combWb = (bf16_t*)alloc((size_t)HH * HE * 2);
  bf16_t* Wihb   = (bf16_t*)alloc((size_t)H3 * HH * 2);
  bf16_t* Whhb   = (bf16_t*)alloc((size_t)H3 * HH * 2);
  bf16_t* catb   = (bf16_t*)alloc((size_t)BB * HE * 2);
  bf16_t* xb     = (bf16_t*)alloc((size_t)BB * HH * 2);
  float*  logits = (float*)alloc((size_t)BB * LL * 4);
  float*  aw     = (float*)alloc((size_t)BB * LL * 4);
  float*  ctx    = (float*)alloc((size_t)BB * HH * 4);
  float*  xf     = (float*)alloc((size_t)BB * HH * 4);
  float*  gi     = (float*)alloc((size_t)BB * H3 * 4);
  float*  gh     = (float*)alloc((size_t)BB * H3 * 4);
  float*  h0     = (float*)alloc((size_t)BB * HH * 4);
  float*  h1     = (float*)alloc((size_t)BB * HH * 4);

  // Optional: bf16 copy of encoder_outputs (128 MB -> fits the 192 MB L2,
  // halving the dominant per-step traffic). Branch depends only on ws_size.
  const size_t enc_bytes = (size_t)BB * LL * HH * 2;
  bf16_t* encb = nullptr;
  if (ws_size >= cur + enc_bytes) encb = (bf16_t*)alloc(enc_bytes);

  // One-time conversions (on stream, graph-capturable)
  auto cvt = [&](const float* s, bf16_t* d, size_t n) {
    size_t n4 = n / 4;
    int blocks = (int)(((n4 + 255) / 256 < 8192) ? (n4 + 255) / 256 : 8192);
    k_f32_to_bf16<<<blocks, 256, 0, stream>>>(s, d, n4);
  };
  cvt(attn_W, attnWb, (size_t)LL * HE);
  cvt(comb_W, combWb, (size_t)HH * HE);
  cvt(W_ih,   Wihb,   (size_t)H3 * HH);
  cvt(W_hh,   Whhb,   (size_t)H3 * HH);
  if (encb) cvt(enc, encb, (size_t)BB * LL * HH);

  hipMemcpyAsync(h0, hidden, (size_t)BB * HH * 4, hipMemcpyDeviceToDevice, stream);

  for (int t = 0; t < TT; ++t) {
    float* hc = (t & 1) ? h1 : h0;
    float* hn = (t & 1) ? h0 : h1;

    // cat = [bf16(emb_t) | bf16(h)]
    k_embed_cat<<<dim3(BB, HE / 256), 256, 0, stream>>>(tgt, emb, hc, catb, t);

    // gh = h @ W_hh.T + b_hh  (uses the h-half of cat before it's overwritten)
    k_wmma_gemm<<<H3 / 128, 256, 0, stream>>>(
        catb + EE, HE, Whhb, HH, b_hh, gh, nullptr, H3, 0);

    // attention logits = cat @ attn_W.T + attn_b
    k_wmma_gemm<<<LL / 128, 256, 0, stream>>>(
        catb, HE, attnWb, HE, attn_b, logits, nullptr, LL, 0);

    // softmax; writes aw (ws) + both attention output copies in d_out
    k_softmax_write<<<BB, 256, 0, stream>>>(logits, aw, out, t);

    // context matvec (L2-resident bf16 path when workspace allows)
    if (encb)
      k_ctx_bf16<<<BB, 128, 0, stream>>>(aw, encb, ctx);
    else
      k_ctx_f32<<<BB, 128, 0, stream>>>(aw, enc, ctx);

    // cat2 = [emb | bf16(ctx)]  (overwrite h-half in place)
    k_ctx_to_cat<<<dim3(BB, HH / 256), 256, 0, stream>>>(ctx, catb);

    // x = relu(cat2 @ comb_W.T + comb_b), also emit bf16 x for next GEMM
    k_wmma_gemm<<<HH / 128, 256, 0, stream>>>(
        catb, HE, combWb, HE, comb_b, xf, xb, HH, 1);

    // gi = x @ W_ih.T + b_ih
    k_wmma_gemm<<<H3 / 128, 256, 0, stream>>>(
        xb, HH, Wihb, HH, b_ih, gi, nullptr, H3, 0);

    // GRU elementwise update
    k_gru<<<dim3(BB, HH / 256), 256, 0, stream>>>(gi, gh, hc, hn);
  }

  // T=128 (even) -> final hidden lives in h0
  hipMemcpyAsync(out + OUT_OFF2, h0, (size_t)BB * HH * 4,
                 hipMemcpyDeviceToDevice, stream);
}
